// _Attention_9594956939409
// MI455X (gfx1250) — matrix operand order repaired and checked by reading, not compiled, not run on hardware
//
#include <hip/hip_runtime.h>

#define DIMC    1024
#define TQ      2048
#define BBATCH  4
#define NHEADS  16
#define DHEAD   64
#define NPATCH  256
#define INNER   1024
#define QKVCOLS 3072

// log2(e) * (1/sqrt(64))
#define KSCALE 0.18033688011112042f

typedef __attribute__((ext_vector_type(16))) _Float16 v16h;
typedef __attribute__((ext_vector_type(8)))  _Float16 v8h;
typedef __attribute__((ext_vector_type(8)))  float    v8f;

union U16x8 { uint4 u; _Float16 h[8]; v8h v; };
union U16x4 { uint2 u; _Float16 h[4]; };

__device__ __forceinline__ v16h combine8(v8h lo, v8h hi) {
  v16h r;
#pragma unroll
  for (int i = 0; i < 8; ++i) { r[i] = lo[i]; r[i + 8] = hi[i]; }
  return r;
}
__device__ __forceinline__ v16h ld_frag16(const _Float16* p0, const _Float16* p1) {
  return combine8(*(const v8h*)p0, *(const v8h*)p1);
}

// CDNA5 async memory->LDS copy (GVS mode): per-lane 16B, tracked by ASYNCcnt.
//   vdst  = LDS byte address (low 32 bits of generic LDS pointer)
//   vaddr = per-lane unsigned byte offset
//   saddr = uniform 64-bit global base
__device__ __forceinline__ void async_load_b128(unsigned lds_addr, unsigned gvoff,
                                                const void* sbase) {
  asm volatile("global_load_async_to_lds_b128 %0, %1, %2"
               :: "v"(lds_addr), "v"(gvoff), "s"(sbase) : "memory");
}
__device__ __forceinline__ void wait_async0() {
  asm volatile("s_wait_asynccnt 0x0" ::: "memory");
}
__device__ __forceinline__ unsigned lds_u32(const void* p) {
  return (unsigned)(unsigned long long)(uintptr_t)p;
}

// ---------------- fused LayerNorm -> f16 ----------------
__global__ __launch_bounds__(256) void k_layernorm_f16(
    const float* __restrict__ x, const float* __restrict__ gamma,
    const float* __restrict__ beta, _Float16* __restrict__ xn) {
  const int row = blockIdx.x;
  const int tid = threadIdx.x;
  const float4 xv = ((const float4*)(x + (size_t)row * DIMC))[tid];
  float s = xv.x + xv.y + xv.z + xv.w;
  float q = xv.x * xv.x + xv.y * xv.y + xv.z * xv.z + xv.w * xv.w;
#pragma unroll
  for (int off = 16; off > 0; off >>= 1) {
    s += __shfl_down(s, off, 32);
    q += __shfl_down(q, off, 32);
  }
  __shared__ float ss[8], sq[8];
  const int wv = tid >> 5, ln = tid & 31;
  if (ln == 0) { ss[wv] = s; sq[wv] = q; }
  __syncthreads();
  if (tid == 0) {
    float S = 0.f, Q = 0.f;
#pragma unroll
    for (int i = 0; i < 8; ++i) { S += ss[i]; Q += sq[i]; }
    float mean = S * (1.0f / DIMC);
    float var  = Q * (1.0f / DIMC) - mean * mean;
    ss[0] = mean;
    sq[0] = rsqrtf(var + 1e-5f);
  }
  __syncthreads();
  const float mean = ss[0], rstd = sq[0];
  const float4 gv = ((const float4*)gamma)[tid];
  const float4 bv = ((const float4*)beta)[tid];
  U16x4 o;
  o.h[0] = (_Float16)((xv.x - mean) * rstd * gv.x + bv.x);
  o.h[1] = (_Float16)((xv.y - mean) * rstd * gv.y + bv.y);
  o.h[2] = (_Float16)((xv.z - mean) * rstd * gv.z + bv.z);
  o.h[3] = (_Float16)((xv.w - mean) * rstd * gv.w + bv.w);
  ((uint2*)(xn + (size_t)row * DIMC))[tid] = o.u;
}

// ---------------- f32 -> f16 weight convert ----------------
__global__ __launch_bounds__(256) void k_f32_to_f16(const float* __restrict__ src,
                                                    _Float16* __restrict__ dst, int n) {
  int i = blockIdx.x * 1024 + threadIdx.x * 4;
  if (i + 3 < n) {
    float4 v = *(const float4*)(src + i);
    U16x4 o;
    o.h[0] = (_Float16)v.x; o.h[1] = (_Float16)v.y;
    o.h[2] = (_Float16)v.z; o.h[3] = (_Float16)v.w;
    *(uint2*)(dst + i) = o.u;
  }
}

// ---------------- QKV GEMM: [8192,1024] x [1024,3072], split to q/k/v [B,H,T,D] f16 --------
__global__ __launch_bounds__(256) void k_gemm_qkv(
    const _Float16* __restrict__ A, const _Float16* __restrict__ W,
    _Float16* __restrict__ q, _Float16* __restrict__ k, _Float16* __restrict__ v) {
  __shared__ _Float16 As[128 * 40];    // [row][k 0..31 +8 pad]
  __shared__ _Float16 BsT[128 * 40];   // [n][k 0..31 +8 pad]  (K-contiguous for B-frags)
  const int tid = threadIdx.x;
  const int mBase = blockIdx.x * 128;
  const int nBase = blockIdx.y * 128;
  const int wv = tid >> 5, lane = tid & 31;
  const int nl = lane & 15, hi = lane >> 4;
  const int mW = (wv >> 2) * 64, nW = (wv & 3) * 32;
  const unsigned ldsA = lds_u32(&As[0]);

  v8f acc[4][2] = {};

  for (int k0 = 0; k0 < DIMC; k0 += 32) {
    __syncthreads();
#pragma unroll
    for (int it = 0; it < 2; ++it) {
      // A tile: async memory->LDS, row-major (128 rows x 32 k)
      int id = wv * 64 + it * 32 + lane;       // 0..511
      int r = id >> 2, seg = id & 3;
      async_load_b128(ldsA + (unsigned)(r * 80 + seg * 16),
                      (unsigned)((((mBase + r) * DIMC) + k0 + seg * 8) * 2), A);
      // W tile: VGPR load + transposed LDS store (K-contiguous per column)
      int id2 = tid + it * 256;
      int kr = id2 >> 4, sg = id2 & 15;
      U16x8 ub;
      ub.u = *(const uint4*)(W + (size_t)(k0 + kr) * QKVCOLS + nBase + sg * 8);
      if (k0 + 32 < DIMC)
        __builtin_prefetch(W + (size_t)(k0 + 32 + kr) * QKVCOLS + nBase + sg * 8, 0, 1);
#pragma unroll
      for (int j = 0; j < 8; ++j) BsT[(sg * 8 + j) * 40 + kr] = ub.h[j];
    }
    wait_async0();
    __syncthreads();

    v16h af[4], bf[2];
#pragma unroll
    for (int mt = 0; mt < 4; ++mt) {
      const _Float16* p = &As[(mW + mt * 16 + nl) * 40];
      af[mt] = ld_frag16(p + hi * 8, p + 16 + hi * 8);
    }
#pragma unroll
    for (int nt = 0; nt < 2; ++nt) {
      const _Float16* p = &BsT[(nW + nt * 16 + nl) * 40 + hi * 8];
      bf[nt] = ld_frag16(p, p + 16);
    }
#pragma unroll
    for (int mt = 0; mt < 4; ++mt)
#pragma unroll
      for (int nt = 0; nt < 2; ++nt)
        acc[mt][nt] = __builtin_amdgcn_wmma_f32_16x16x32_f16(
            false, af[mt], false, bf[nt], (short)0, acc[mt][nt], false, false);
  }

#pragma unroll
  for (int mt = 0; mt < 4; ++mt)
#pragma unroll
    for (int nt = 0; nt < 2; ++nt)
#pragma unroll
      for (int r = 0; r < 8; ++r) {
        int m = mBase + mW + mt * 16 + r + hi * 8;
        int c = nBase + nW + nt * 16 + nl;
        int which = c >> 10, rem = c & 1023;
        int hh = rem >> 6, d = rem & 63;
        int b = m >> 11, t = m & 2047;
        _Float16* dst = which == 0 ? q : (which == 1 ? k : v);
        dst[(((size_t)(b * NHEADS + hh)) * TQ + t) * DHEAD + d] =
            (_Float16)acc[mt][nt][r];
      }
}

// ---------------- fused flash attention (block-causal, uniform per tile) ----------------
__global__ __launch_bounds__(256) void k_attention(
    const _Float16* __restrict__ qg, const _Float16* __restrict__ kg,
    const _Float16* __restrict__ vg, _Float16* __restrict__ og) {
  // Ks view [128 key][64 d + 8 pad] overlaps per-wave P tiles [16][128 + 8 pad]
  __shared__ _Float16 sKP[8 * 16 * 136];   // 34816 B
  __shared__ _Float16 sVT[64 * 136];       // V transposed: [d][key + pad], 17408 B
  const int tid = threadIdx.x;
  const int wv = tid >> 5, lane = tid & 31;
  const int nl = lane & 15, hi = lane >> 4, hi8 = hi * 8;
  const int h = blockIdx.y, b = blockIdx.z;
  const int qStart = blockIdx.x * 128;
  const int Lkeys = (qStart / NPATCH + 1) * NPATCH;   // uniform mask -> key range only
  const size_t headBase = ((size_t)(b * NHEADS + h)) * TQ * DHEAD;
  const unsigned ldsK = lds_u32(&sKP[0]);
  const _Float16* kbase = kg + headBase;

  // preload Q fragments (16 rows per wave, full D=64 as two K=32 chunks)
  const int tq0 = qStart + wv * 16 + nl;
  const uint4* qrow = (const uint4*)(qg + headBase + (size_t)tq0 * DHEAD);
  v16h qf[2];
#pragma unroll
  for (int kc = 0; kc < 2; ++kc) {
    U16x8 u0, u1;
    u0.u = qrow[kc * 4 + hi];
    u1.u = qrow[kc * 4 + 2 + hi];
    qf[kc] = combine8(u0.v, u1.v);
  }

  v8f O[4] = {};
  float mrow[8], lrow[8];
#pragma unroll
  for (int r = 0; r < 8; ++r) { mrow[r] = -__builtin_huge_valf(); lrow[r] = 0.f; }

  _Float16* Pw = sKP + wv * (16 * 136);

  for (int kt = 0; kt < Lkeys; kt += 128) {
    // K tile: async memory->LDS (row-major). V tile: VGPR + transposed LDS store.
#pragma unroll
    for (int i = 0; i < 4; ++i) {
      int id = wv * 128 + i * 32 + lane;     // 0..1023
      int key = id >> 3, seg = id & 7;
      async_load_b128(ldsK + (unsigned)(key * 144 + seg * 16),
                      (unsigned)((((kt + key) * DHEAD) + seg * 8) * 2), kbase);
      U16x8 uv;
      uv.u = *(const uint4*)(vg + headBase + (size_t)(kt + key) * DHEAD + seg * 8);
#pragma unroll
      for (int j = 0; j < 8; ++j) sVT[(seg * 8 + j) * 136 + key] = uv.h[j];
    }
    wait_async0();
    __syncthreads();

    // S = Q x K^T  (16 x 128 per wave)
    v8f S[8] = {};
#pragma unroll
    for (int kc = 0; kc < 2; ++kc)
#pragma unroll
      for (int nt = 0; nt < 8; ++nt) {
        const _Float16* p = &sKP[(nt * 16 + nl) * 72 + kc * 32 + hi * 8];
        v16h bf = ld_frag16(p, p + 16);
        S[nt] = __builtin_amdgcn_wmma_f32_16x16x32_f16(
            false, qf[kc], false, bf, (short)0, S[nt], false, false);
      }
    __syncthreads();   // all waves done reading Ks before P overwrites the region

    // online softmax: row stats via shfl within each 16-lane half, write P (f16)
#pragma unroll
    for (int r = 0; r < 8; ++r) {
      float mx = S[0][r];
#pragma unroll
      for (int nt = 1; nt < 8; ++nt) mx = fmaxf(mx, S[nt][r]);
#pragma unroll
      for (int msk = 8; msk > 0; msk >>= 1) mx = fmaxf(mx, __shfl_xor(mx, msk, 32));
      float mNew = fmaxf(mrow[r], mx);
      float corr = exp2f((mrow[r] - mNew) * KSCALE);
      mrow[r] = mNew;
      float rs = 0.f;
#pragma unroll
      for (int nt = 0; nt < 8; ++nt) {
        float p = exp2f((S[nt][r] - mNew) * KSCALE);
        rs += p;
        Pw[(r + hi8) * 136 + nt * 16 + nl] = (_Float16)p;
      }
#pragma unroll
      for (int msk = 8; msk > 0; msk >>= 1) rs += __shfl_xor(rs, msk, 32);
      lrow[r] = lrow[r] * corr + rs;
#pragma unroll
      for (int nt4 = 0; nt4 < 4; ++nt4) O[nt4][r] *= corr;
    }
    __syncthreads();

    // O += P x V
#pragma unroll
    for (int kc2 = 0; kc2 < 4; ++kc2) {
      const _Float16* ap = &Pw[nl * 136 + kc2 * 32 + hi8];
      v16h af = ld_frag16(ap, ap + 16);
#pragma unroll
      for (int nt4 = 0; nt4 < 4; ++nt4) {
        const _Float16* bp = &sVT[(nt4 * 16 + nl) * 136 + kc2 * 32 + hi * 8];
        v16h bf = ld_frag16(bp, bp + 16);
        O[nt4] = __builtin_amdgcn_wmma_f32_16x16x32_f16(
            false, af, false, bf, (short)0, O[nt4], false, false);
      }
    }
    __syncthreads();   // done with P/VsT before next tile's loads
  }

  // normalize and store merged heads [B, T, H*D] f16
#pragma unroll
  for (int nt4 = 0; nt4 < 4; ++nt4)
#pragma unroll
    for (int r = 0; r < 8; ++r) {
      int tq = qStart + wv * 16 + r + hi8;
      int d = nt4 * 16 + nl;
      og[((size_t)b * TQ + tq) * INNER + h * DHEAD + d] =
          (_Float16)(O[nt4][r] / lrow[r]);
    }
}

// ---------------- output projection: [8192,1024] x [1024,1024] + bias -> f32 ----------------
__global__ __launch_bounds__(256) void k_gemm_out(
    const _Float16* __restrict__ A, const _Float16* __restrict__ W,
    const float* __restrict__ bias, float* __restrict__ out) {
  __shared__ _Float16 As[128 * 40];
  __shared__ _Float16 BsT[128 * 40];
  const int tid = threadIdx.x;
  const int mBase = blockIdx.x * 128;
  const int nBase = blockIdx.y * 128;
  const int wv = tid >> 5, lane = tid & 31;
  const int nl = lane & 15, hi = lane >> 4;
  const int mW = (wv >> 2) * 64, nW = (wv & 3) * 32;
  const unsigned ldsA = lds_u32(&As[0]);

  v8f acc[4][2] = {};

  for (int k0 = 0; k0 < DIMC; k0 += 32) {
    __syncthreads();
#pragma unroll
    for (int it = 0; it < 2; ++it) {
      int id = wv * 64 + it * 32 + lane;
      int r = id >> 2, seg = id & 3;
      async_load_b128(ldsA + (unsigned)(r * 80 + seg * 16),
                      (unsigned)((((mBase + r) * DIMC) + k0 + seg * 8) * 2), A);
      int id2 = tid + it * 256;
      int kr = id2 >> 4, sg = id2 & 15;
      U16x8 ub;
      ub.u = *(const uint4*)(W + (size_t)(k0 + kr) * DIMC + nBase + sg * 8);
      if (k0 + 32 < DIMC)
        __builtin_prefetch(W + (size_t)(k0 + 32 + kr) * DIMC + nBase + sg * 8, 0, 1);
#pragma unroll
      for (int j = 0; j < 8; ++j) BsT[(sg * 8 + j) * 40 + kr] = ub.h[j];
    }
    wait_async0();
    __syncthreads();

    v16h af[4], bf[2];
#pragma unroll
    for (int mt = 0; mt < 4; ++mt) {
      const _Float16* p = &As[(mW + mt * 16 + nl) * 40];
      af[mt] = ld_frag16(p + hi * 8, p + 16 + hi * 8);
    }
#pragma unroll
    for (int nt = 0; nt < 2; ++nt) {
      const _Float16* p = &BsT[(nW + nt * 16 + nl) * 40 + hi * 8];
      bf[nt] = ld_frag16(p, p + 16);
    }
#pragma unroll
    for (int mt = 0; mt < 4; ++mt)
#pragma unroll
      for (int nt = 0; nt < 2; ++nt)
        acc[mt][nt] = __builtin_amdgcn_wmma_f32_16x16x32_f16(
            false, af[mt], false, bf[nt], (short)0, acc[mt][nt], false, false);
  }

#pragma unroll
  for (int mt = 0; mt < 4; ++mt)
#pragma unroll
    for (int nt = 0; nt < 2; ++nt)
#pragma unroll
      for (int r = 0; r < 8; ++r) {
        int m = mBase + mW + mt * 16 + r + hi * 8;
        int c = nBase + nW + nt * 16 + nl;
        out[(size_t)m * DIMC + c] = acc[mt][nt][r] + bias[c];
      }
}

extern "C" void kernel_launch(void* const* d_in, const int* in_sizes, int n_in,
                              void* d_out, int out_size, void* d_ws, size_t ws_size,
                              hipStream_t stream) {
  (void)in_sizes; (void)n_in; (void)out_size; (void)ws_size;
  const float* x     = (const float*)d_in[0];
  const float* gamma = (const float*)d_in[1];
  const float* beta  = (const float*)d_in[2];
  const float* wqkv  = (const float*)d_in[3];
  const float* wout  = (const float*)d_in[4];
  const float* bout  = (const float*)d_in[5];
  float* out = (float*)d_out;

  char* ws = (char*)d_ws;
  size_t off = 0;
  auto carve = [&](size_t bytes) -> char* {
    char* p = ws + off;
    off += (bytes + 255) & ~(size_t)255;
    return p;
  };
  _Float16* xn_h   = (_Float16*)carve((size_t)BBATCH * TQ * DIMC * 2);
  _Float16* wqkv_h = (_Float16*)carve((size_t)DIMC * QKVCOLS * 2);
  _Float16* wout_h = (_Float16*)carve((size_t)DIMC * DIMC * 2);
  _Float16* q_h    = (_Float16*)carve((size_t)BBATCH * NHEADS * TQ * DHEAD * 2);
  _Float16* k_h    = (_Float16*)carve((size_t)BBATCH * NHEADS * TQ * DHEAD * 2);
  _Float16* v_h    = (_Float16*)carve((size_t)BBATCH * NHEADS * TQ * DHEAD * 2);
  _Float16* o_h    = (_Float16*)carve((size_t)BBATCH * TQ * INNER * 2);

  k_f32_to_f16<<<(DIMC * QKVCOLS) / 1024, 256, 0, stream>>>(wqkv, wqkv_h, DIMC * QKVCOLS);
  k_f32_to_f16<<<(DIMC * DIMC) / 1024, 256, 0, stream>>>(wout, wout_h, DIMC * DIMC);
  k_layernorm_f16<<<BBATCH * TQ, 256, 0, stream>>>(x, gamma, beta, xn_h);
  k_gemm_qkv<<<dim3(64, 24), 256, 0, stream>>>(xn_h, wqkv_h, q_h, k_h, v_h);
  k_attention<<<dim3(TQ / 128, NHEADS, BBATCH), 256, 0, stream>>>(q_h, k_h, v_h, o_h);
  k_gemm_out<<<dim3(64, 8), 256, 0, stream>>>(o_h, wout_h, bout, out);
}
